// Attention_44581760533021
// MI455X (gfx1250) — matrix operand order repaired and checked by reading, not compiled, not run on hardware
//
#include <hip/hip_runtime.h>

typedef unsigned short u16;
typedef unsigned int   u32;
typedef unsigned long long u64;
typedef __attribute__((ext_vector_type(16))) __bf16 v16bf;
typedef __attribute__((ext_vector_type(8)))  float  v8f;
typedef __attribute__((ext_vector_type(4)))  u32    v4u;
typedef __attribute__((ext_vector_type(4)))  u32    u32x4;
typedef __attribute__((ext_vector_type(8)))  int    i32x8;
typedef __attribute__((ext_vector_type(4)))  int    i32x4;

union Frag {
  v16bf v;
  v4u   q[2];
};

static __device__ inline v8f vzero() {
  v8f z;
#pragma unroll
  for (int i = 0; i < 8; ++i) z[i] = 0.0f;
  return z;
}

// f32 -> bf16 round-to-nearest-even
static __device__ inline u16 f2bf(float f) {
  union { float f; u32 u; } cv; cv.f = f;
  u32 u = cv.u;
  return (u16)((u + 0x7FFFu + ((u >> 16) & 1u)) >> 16);
}

static __device__ inline v8f wmma_bf16(const Frag& a, const Frag& b, v8f c) {
  return __builtin_amdgcn_wmma_f32_16x16x32_bf16(false, a.v, false, b.v,
                                                 (short)0, c, false, false);
}

// ---------------------------------------------------------------------------
// Tensor Data Mover: DMA a (up to 2-D) bf16 tile from global into LDS.
// D# packing per CDNA5 ISA:
//   group0: [1:0]=count(1), [63:32]=lds_addr, [120:64]=global_addr, [127:126]=2
//   group1: [17:16]=data_size(1 => 2B), [79:48]=tensor_dim0, [111:80]=tensor_dim1,
//           [127:112]=tile_dim0, [143:128]=tile_dim1, [207:160]=tensor_dim0_stride
// Dims/strides are in data_size (element) units.
// This toolchain exposes the 6-arg builtin (extra int32x8 group before cpol).
// ---------------------------------------------------------------------------
static __device__ inline void tdm_load_to_lds(u32 lds_addr, const void* gptr,
                                              u32 tensor_d0, u32 tensor_d1,
                                              u32 tile_d0, u32 tile_d1,
                                              u64 stride0) {
  const u64 ga = (u64)(size_t)gptr;
  u32x4 g0;
  g0.x = 1u;                                            // count=1, user-mode
  g0.y = lds_addr;                                      // LDS byte address
  g0.z = (u32)ga;                                       // global addr [95:64]
  g0.w = (u32)((ga >> 32) & 0x01FFFFFFu) | (2u << 30);  // addr[56:32] | type=2
  i32x8 g1;
  g1[0] = (int)(1u << 16);                              // data_size = 2 bytes
  g1[1] = (int)((tensor_d0 & 0xFFFFu) << 16);
  g1[2] = (int)((tensor_d0 >> 16) | ((tensor_d1 & 0xFFFFu) << 16));
  g1[3] = (int)((tensor_d1 >> 16) | (tile_d0 << 16));
  g1[4] = (int)tile_d1;                                 // tile_dim2 = 0
  g1[5] = (int)(u32)stride0;                            // stride0 [31:0]
  g1[6] = (int)(u32)((stride0 >> 32) & 0xFFFFu);        // stride0 [47:32]
  g1[7] = 0;
  const i32x4 z4 = {0, 0, 0, 0};                        // groups 2/3 unused (2-D)
  const i32x8 z8 = {0, 0, 0, 0, 0, 0, 0, 0};
  __builtin_amdgcn_tensor_load_to_lds(g0, g1, z4, z4, z8, 0);
}

// ---------------------------------------------------------------------------
// Core: one wave computes a 16(M) x 64(N) strip of C = A(bf16) * Bt(bf16)^T
// A: row-major [M][K], Bt: row-major [N][K] (i.e. B transposed).
// ---------------------------------------------------------------------------
static __device__ inline void gemm16x64(const u16* __restrict__ A, int lda,
                                        const u16* __restrict__ Bt, int ldb,
                                        int K, int lane, v8f acc[4]) {
  const int row  = lane & 15;
  const int half = lane >> 4;
  const u16* ap0 = A + (size_t)row * lda;
  for (int kk = 0; kk < K; kk += 32) {
    Frag a;
    const u16* ap = ap0 + kk + half * 16;
    a.q[0] = *(const v4u*)(ap);        // K = kk + half*16 + [0..7]
    a.q[1] = *(const v4u*)(ap + 8);    // K = kk + half*16 + [8..15]
#pragma unroll
    for (int t = 0; t < 4; ++t) {
      Frag b;
      const u16* bp = Bt + (size_t)(t * 16 + row) * ldb + kk + half * 16;
      b.q[0] = *(const v4u*)(bp);      // K = kk + half*16 + [0..7]
      b.q[1] = *(const v4u*)(bp + 8);  // K = kk + half*16 + [8..15]
      acc[t] = wmma_bf16(a, b, acc[t]);
    }
  }
}

// ---------------------------------------------------------------------------
// Prep: elementwise f32 -> bf16 (packed 2/thread)
// ---------------------------------------------------------------------------
__global__ __launch_bounds__(256) void cvt_bf16_kernel(const float* __restrict__ in,
                                                       u16* __restrict__ out, int n2) {
  int i = blockIdx.x * blockDim.x + threadIdx.x;
  if (i < n2) {
    float a = in[2 * i], b = in[2 * i + 1];
    ((u32*)out)[i] = (u32)f2bf(a) | ((u32)f2bf(b) << 16);
  }
}

// ---------------------------------------------------------------------------
// Prep: transpose f32 [R][C] -> bf16 [C][R] (32x32 LDS tiles)
// ---------------------------------------------------------------------------
__global__ __launch_bounds__(256) void transpose_cvt(const float* __restrict__ in,
                                                     u16* __restrict__ out,
                                                     int R, int C) {
  __shared__ float tile[32][33];
  int x  = blockIdx.x * 32 + threadIdx.x;
  int y0 = blockIdx.y * 32 + threadIdx.y;
#pragma unroll
  for (int j = 0; j < 4; ++j)
    tile[threadIdx.y + j * 8][threadIdx.x] = in[(size_t)(y0 + j * 8) * C + x];
  __syncthreads();
  int ox  = blockIdx.y * 32 + threadIdx.x;
  int oy0 = blockIdx.x * 32 + threadIdx.y;
#pragma unroll
  for (int j = 0; j < 4; ++j)
    out[(size_t)(oy0 + j * 8) * R + ox] = f2bf(tile[threadIdx.x][threadIdx.y + j * 8]);
}

// ---------------------------------------------------------------------------
// Kernel 1: fused QKV projection.
// grid = (3072/64, 4096/128), block 256 (8 waves, 16 rows each).
// Writes Q [B,H,N,64] ; K (pre-scaled 1/8) [B,H,N,64] ; V transposed [B,H,64,N].
// ---------------------------------------------------------------------------
__global__ __launch_bounds__(256) void qkv_kernel(const u16* __restrict__ xb,
                                                  const u16* __restrict__ wqkt,
                                                  const u16* __restrict__ wvt,
                                                  u16* __restrict__ qb,
                                                  u16* __restrict__ kbuf,
                                                  u16* __restrict__ vtb) {
  const int lane = threadIdx.x & 31;
  const int wv   = threadIdx.x >> 5;
  const int c0   = blockIdx.x * 64;
  const int m0   = blockIdx.y * 128 + wv * 16;
  const u16* Bt  = (c0 < 2048) ? (wqkt + (size_t)c0 * 1024)
                               : (wvt + (size_t)(c0 - 2048) * 1024);
  v8f acc[4] = {vzero(), vzero(), vzero(), vzero()};
  gemm16x64(xb + (size_t)m0 * 1024, 1024, Bt, 1024, 1024, lane, acc);

  const int row = lane & 15, half = lane >> 4;
#pragma unroll
  for (int t = 0; t < 4; ++t) {
#pragma unroll
    for (int r = 0; r < 8; ++r) {
      const int mm = m0 + r + 8 * half;      // flat token row
      const int c  = c0 + t * 16 + row;      // global output column
      const int b  = mm >> 11, n = mm & 2047;
      const float val = acc[t][r];
      if (c < 2048) {                        // qk half: c = sel*1024 + h*64 + d
        const int h = (c >> 6) & 15, d = c & 63;
        const size_t off = ((size_t)(b * 16 + h) * 2048 + n) * 64 + d;
        if (c < 1024) qb[off] = f2bf(val);
        else          kbuf[off] = f2bf(val * 0.125f);   // 1/sqrt(64)
      } else {                               // v half
        const int cv2 = c - 2048;
        const int h = cv2 >> 6, d = cv2 & 63;
        vtb[((size_t)(b * 16 + h) * 64 + d) * 2048 + n] = f2bf(val);
      }
    }
  }
}

// ---------------------------------------------------------------------------
// Kernel 2: flash attention. grid = (2048/128, B*H), block 256 (8 waves).
// K/V chunks are double-buffered into LDS by the Tensor Data Mover (one wave
// issues, TENSORcnt + workgroup barrier synchronize); all 8 waves consume the
// shared LDS copy, eliminating the 8x redundant global fetch of K/V.
// ---------------------------------------------------------------------------
__global__ __launch_bounds__(256) void attn_kernel(const u16* __restrict__ qb,
                                                   const u16* __restrict__ kbuf,
                                                   const u16* __restrict__ vtb,
                                                   u16* __restrict__ ob) {
  __shared__ u16 klds[2][32 * 64];   // [buf][key-row][d]        8 KB
  __shared__ u16 vlds[2][64 * 32];   // [buf][d-row][key]        8 KB
  __shared__ u16 pmem[8 * 512];      // per-wave 16x32 P staging 8 KB

  const int lane = threadIdx.x & 31;
  const int wv   = threadIdx.x >> 5;
  const int row  = lane & 15, half = lane >> 4;
  const int bh   = blockIdx.y;
  const int q0   = (blockIdx.x * 8 + wv) * 16;
  const u16* Qp  = qb   + (size_t)bh * 2048 * 64;
  const u16* Kp  = kbuf + (size_t)bh * 2048 * 64;
  const u16* Vp  = vtb  + (size_t)bh * 64 * 2048;   // [64][2048]
  u16* pl = pmem + wv * 512;

  auto issue_chunk = [&](int n0, int buf) {
    // K chunk: 32 rows x 64 d, contiguous => 1-D tile of 2048 elems (4 KB)
    tdm_load_to_lds((u32)(size_t)(void*)&klds[buf][0], Kp + (size_t)n0 * 64,
                    131072u, 0u, 2048u, 0u, 0ull);
    // V^T chunk: 64 rows (d) x 32 keys, row stride 2048 elems
    tdm_load_to_lds((u32)(size_t)(void*)&vlds[buf][0], Vp + n0,
                    2048u, 64u, 32u, 64u, 2048ull);
  };

  if (wv == 0) issue_chunk(0, 0);   // preload chunk 0

  // Q A-fragments (two K-steps of 32) live in registers for the whole loop.
  Frag aQ0, aQ1;
  {
    const u16* qbase = Qp + (size_t)(q0 + row) * 64;
    aQ0.q[0] = *(const v4u*)(qbase + half * 16);
    aQ0.q[1] = *(const v4u*)(qbase + 8 + half * 16);
    aQ1.q[0] = *(const v4u*)(qbase + 32 + half * 16);
    aQ1.q[1] = *(const v4u*)(qbase + 40 + half * 16);
  }

  float m[8], ssum[8];
  v8f o0 = vzero(), o1 = vzero(), o2 = vzero(), o3 = vzero();
#pragma unroll
  for (int r = 0; r < 8; ++r) { m[r] = -1e30f; ssum[r] = 0.0f; }

  for (int i = 0; i < 64; ++i) {
    const int cur = i & 1;
    // chunk i was issued last iteration (or in the prologue); wait for the DMA
    if (wv == 0) __builtin_amdgcn_s_wait_tensorcnt(0);
    __syncthreads();   // buf `cur` ready for all waves; all waves are done
                       // reading buf `cur^1` (their chunk i-1 WMMAs consumed it)
    if (wv == 0 && i + 1 < 64) issue_chunk((i + 1) * 32, cur ^ 1);

    // scores: 16 queries x 32 keys, contraction over head_dim=64 (from LDS)
    v8f s0 = vzero(), s1 = vzero();
    {
      Frag b;
      const u16* kb0 = &klds[cur][(size_t)row * 64 + half * 16];
      b.q[0] = *(const v4u*)(kb0);
      b.q[1] = *(const v4u*)(kb0 + 8);
      s0 = wmma_bf16(aQ0, b, s0);
      b.q[0] = *(const v4u*)(kb0 + 32);
      b.q[1] = *(const v4u*)(kb0 + 40);
      s0 = wmma_bf16(aQ1, b, s0);
      const u16* kb1 = kb0 + 16 * 64;
      b.q[0] = *(const v4u*)(kb1);
      b.q[1] = *(const v4u*)(kb1 + 8);
      s1 = wmma_bf16(aQ0, b, s1);
      b.q[0] = *(const v4u*)(kb1 + 32);
      b.q[1] = *(const v4u*)(kb1 + 40);
      s1 = wmma_bf16(aQ1, b, s1);
    }

    // online softmax: C-layout rows are M = r + 8*half striped over the 16
    // lanes of each half, so xor-masks 1/2/4/8 reduce exactly one row.
    float alpha[8];
#pragma unroll
    for (int r = 0; r < 8; ++r) {
      float cm = fmaxf(s0[r], s1[r]);
      cm = fmaxf(cm, __shfl_xor(cm, 1));
      cm = fmaxf(cm, __shfl_xor(cm, 2));
      cm = fmaxf(cm, __shfl_xor(cm, 4));
      cm = fmaxf(cm, __shfl_xor(cm, 8));
      const float mn = fmaxf(m[r], cm);
      alpha[r] = __expf(m[r] - mn);
      m[r] = mn;
      const float p0 = __expf(s0[r] - mn);
      const float p1 = __expf(s1[r] - mn);
      float rs = p0 + p1;
      rs += __shfl_xor(rs, 1);
      rs += __shfl_xor(rs, 2);
      rs += __shfl_xor(rs, 4);
      rs += __shfl_xor(rs, 8);
      ssum[r] = ssum[r] * alpha[r] + rs;
      // stage P (C-layout) into LDS row-major [16 q rows][32 key cols]
      pl[(r + 8 * half) * 32 + row]      = f2bf(p0);
      pl[(r + 8 * half) * 32 + 16 + row] = f2bf(p1);
    }
#pragma unroll
    for (int r = 0; r < 8; ++r) {
      o0[r] *= alpha[r]; o1[r] *= alpha[r]; o2[r] *= alpha[r]; o3[r] *= alpha[r];
    }

    // reload P as an A-fragment (16x32) via ds_read_b128
    Frag aP;
    const u16* pr = pl + row * 32 + half * 16;
    aP.q[0] = *(const v4u*)(pr);
    aP.q[1] = *(const v4u*)(pr + 8);

    // O += P @ V_chunk ; V^T rows (contiguous keys) now come from LDS
    Frag b;
    const u16* vb = &vlds[cur][(size_t)row * 32 + half * 16];
    b.q[0] = *(const v4u*)(vb);
    b.q[1] = *(const v4u*)(vb + 8);
    o0 = wmma_bf16(aP, b, o0);
    b.q[0] = *(const v4u*)(vb + 16 * 32);
    b.q[1] = *(const v4u*)(vb + 16 * 32 + 8);
    o1 = wmma_bf16(aP, b, o1);
    b.q[0] = *(const v4u*)(vb + 32 * 32);
    b.q[1] = *(const v4u*)(vb + 32 * 32 + 8);
    o2 = wmma_bf16(aP, b, o2);
    b.q[0] = *(const v4u*)(vb + 48 * 32);
    b.q[1] = *(const v4u*)(vb + 48 * 32 + 8);
    o3 = wmma_bf16(aP, b, o3);
  }

  // normalize + write [B, N, H*64] bf16
  const int bb = bh >> 4, h = bh & 15;
#pragma unroll
  for (int r = 0; r < 8; ++r) {
    const float inv = 1.0f / ssum[r];
    const size_t base =
        ((size_t)(bb * 2048 + q0 + r + 8 * half)) * 1024 + h * 64 + row;
    ob[base]      = f2bf(o0[r] * inv);
    ob[base + 16] = f2bf(o1[r] * inv);
    ob[base + 32] = f2bf(o2[r] * inv);
    ob[base + 48] = f2bf(o3[r] * inv);
  }
}

// ---------------------------------------------------------------------------
// Kernel 3: out = O @ w_proj + b_proj, f32 output.
// grid = (1024/64, 4096/128), block 256.
// ---------------------------------------------------------------------------
__global__ __launch_bounds__(256) void proj_kernel(const u16* __restrict__ ob,
                                                   const u16* __restrict__ wpt,
                                                   const float* __restrict__ bias,
                                                   float* __restrict__ out) {
  const int lane = threadIdx.x & 31;
  const int wv   = threadIdx.x >> 5;
  const int c0   = blockIdx.x * 64;
  const int m0   = blockIdx.y * 128 + wv * 16;
  v8f acc[4] = {vzero(), vzero(), vzero(), vzero()};
  gemm16x64(ob + (size_t)m0 * 1024, 1024, wpt + (size_t)c0 * 1024, 1024, 1024,
            lane, acc);
  const int row = lane & 15, half = lane >> 4;
#pragma unroll
  for (int t = 0; t < 4; ++t) {
    const float bcol = bias[c0 + t * 16 + row];
#pragma unroll
    for (int r = 0; r < 8; ++r) {
      const int mm = m0 + r + 8 * half;
      out[(size_t)mm * 1024 + c0 + t * 16 + row] = acc[t][r] + bcol;
    }
  }
}

// ---------------------------------------------------------------------------
extern "C" void kernel_launch(void* const* d_in, const int* in_sizes, int n_in,
                              void* d_out, int out_size, void* d_ws, size_t ws_size,
                              hipStream_t stream) {
  (void)in_sizes; (void)n_in; (void)out_size; (void)ws_size;
  const float* x      = (const float*)d_in[0];  // [2,2048,1024]
  const float* w_qk   = (const float*)d_in[1];  // [1024,2048]
  const float* w_v    = (const float*)d_in[2];  // [1024,1024]
  const float* w_proj = (const float*)d_in[3];  // [1024,1024]
  const float* b_proj = (const float*)d_in[4];  // [1024]
  float* out = (float*)d_out;

  char* ws = (char*)d_ws;
  const size_t MB = 1024 * 1024;
  u16* xb    = (u16*)(ws + 0 * MB);   // [4096][1024] bf16      (8 MB)
  u16* wqkt  = (u16*)(ws + 8 * MB);   // [2048][1024] bf16      (4 MB)
  u16* wvt   = (u16*)(ws + 12 * MB);  // [1024][1024] bf16      (2 MB)
  u16* wpt   = (u16*)(ws + 14 * MB);  // [1024][1024] bf16      (2 MB)
  u16* qb    = (u16*)(ws + 16 * MB);  // [B,H,2048,64] bf16     (8 MB)
  u16* kbuf  = (u16*)(ws + 24 * MB);  // [B,H,2048,64] bf16     (8 MB)
  u16* vtb   = (u16*)(ws + 32 * MB);  // [B,H,64,2048] bf16     (8 MB)
  u16* obuf  = (u16*)(ws + 40 * MB);  // [4096][1024] bf16      (8 MB)

  // prep
  cvt_bf16_kernel<<<(4096 * 1024 / 2 + 255) / 256, 256, 0, stream>>>(
      x, xb, 4096 * 1024 / 2);
  transpose_cvt<<<dim3(2048 / 32, 1024 / 32), dim3(32, 8), 0, stream>>>(
      w_qk, wqkt, 1024, 2048);
  transpose_cvt<<<dim3(1024 / 32, 1024 / 32), dim3(32, 8), 0, stream>>>(
      w_v, wvt, 1024, 1024);
  transpose_cvt<<<dim3(1024 / 32, 1024 / 32), dim3(32, 8), 0, stream>>>(
      w_proj, wpt, 1024, 1024);

  // QKV projection (N_total = 2048 qk cols + 1024 v cols)
  qkv_kernel<<<dim3(3072 / 64, 4096 / 128), 256, 0, stream>>>(
      xb, wqkt, wvt, qb, kbuf, vtb);

  // flash attention (TDM-staged K/V)
  attn_kernel<<<dim3(2048 / 128, 2 * 16), 256, 0, stream>>>(qb, kbuf, vtb, obuf);

  // output projection + bias
  proj_kernel<<<dim3(1024 / 64, 4096 / 128), 256, 0, stream>>>(
      obuf, wpt, b_proj, out);
}
